// FEM_14757507629401
// MI455X (gfx1250) — matrix operand order repaired and checked by reading, not compiled, not run on hardware
//
#include <hip/hip_runtime.h>

// ---------------------------------------------------------------------------
// Graph diffusion conv (Graph-WaveNet gconv) for MI455X / gfx1250.
// bf16 WMMA (v_wmma_f32_16x16x32_bf16) with fp32 accumulation.
// Three phases through a bf16 workspace:
//   prep:     pad/transpose adjacencies + weights, x -> bf16
//   diffuse:  y1 = A^T x, y2 = A^T y1  (8 chains, LDS-resident y1)
//   contract: out = W @ [x|y...] + b + residual  (batched GEMM)
// Workspace: 3,145,728 (adj) + 262,144 (W) + 86,016,000 (xbf) + 688,128,000
//            (h slots) = 777,551,872 bytes.
// ---------------------------------------------------------------------------

typedef __attribute__((ext_vector_type(16))) __bf16 v16bf;
typedef __attribute__((ext_vector_type(8)))  float  v8f;
typedef __attribute__((ext_vector_type(4)))  unsigned int uint4v;

#define XSZ      21504000   // B*C*V*L elements (one tensor)
#define NSLICE   336000     // C*V*L = 56*6000
#define CVL      6000       // V*L
#define ADJ_EL   (512*512)  // one padded adjacency (bf16 elems)
#define WPAD_EL  (64*512)   // one padded weight (bf16 elems)

__device__ __forceinline__ unsigned short f2bf(float f) {
  union { float f; unsigned u; } v; v.f = f;
  unsigned r = v.u + 0x7FFFu + ((v.u >> 16) & 1u);   // round-to-nearest-even
  return (unsigned short)(r >> 16);
}

union FragB16 { v16bf v; uint4v q[2]; };

// ---------------------------------------------------------------------------
// prep_adj: 6 matrices, 512x512 bf16, zero padded.
//   mats 0..4 = transpose(adp0, adp1, ii, io, oo)  (A-frag wants AdjT[w][v])
//   mat  5    = io as-is (for the io^T chain)
// ---------------------------------------------------------------------------
__global__ __launch_bounds__(256) void prep_adj(
    const float* __restrict__ a0, const float* __restrict__ a1,
    const float* __restrict__ a2, const float* __restrict__ a3,
    const float* __restrict__ a4, unsigned short* __restrict__ dst) {
  int idx = blockIdx.x * 256 + threadIdx.x;          // 6*262144 total
  int mat = idx >> 18;
  int rr  = idx & (ADJ_EL - 1);
  int w = rr >> 9, v = rr & 511;
  const float* src;
  switch (mat) {
    case 0: src = a0; break; case 1: src = a1; break; case 2: src = a2; break;
    case 3: src = a3; break; case 4: src = a4; break; default: src = a3; break;
  }
  float val = 0.f;
  if (w < 500 && v < 500) val = (mat < 5) ? src[v * 500 + w] : src[w * 500 + v];
  dst[idx] = f2bf(val);
}

// ---------------------------------------------------------------------------
// prep_w: 4 weights [56,392] -> bf16 [64][7*64 -> 512] zero padded per block.
//   order: 0=w_ii, 1=w_io, 2=w_oo, 3=w_oi  (matches output ids)
// ---------------------------------------------------------------------------
__global__ __launch_bounds__(256) void prep_w(
    const float* __restrict__ w_ii, const float* __restrict__ w_io,
    const float* __restrict__ w_oo, const float* __restrict__ w_oi,
    unsigned short* __restrict__ dst) {
  int idx = blockIdx.x * 256 + threadIdx.x;          // 4*64*512
  int mtx = idx >> 15;
  int rr  = idx & 32767;
  int o = rr >> 9, k = rr & 511;
  int blk = k >> 6, kk = k & 63;
  const float* w = (mtx == 0) ? w_ii : ((mtx == 1) ? w_io : ((mtx == 2) ? w_oo : w_oi));
  float val = 0.f;
  if (o < 56 && blk < 7 && kk < 56) val = w[o * 392 + blk * 56 + kk];
  dst[idx] = f2bf(val);
}

// ---------------------------------------------------------------------------
// prep_x: inflow/outflow fp32 -> bf16 (identity h-block for contraction)
// ---------------------------------------------------------------------------
__global__ __launch_bounds__(256) void prep_x(
    const float* __restrict__ inflow, const float* __restrict__ outflow,
    unsigned short* __restrict__ xbf) {
  long long idx = (long long)blockIdx.x * 256 + threadIdx.x;
  if (idx >= 2LL * XSZ) return;
  float v = (idx < XSZ) ? inflow[idx] : outflow[idx - XSZ];
  xbf[idx] = f2bf(v);
}

// ---------------------------------------------------------------------------
// diffuse: workgroup = (n, chain, channel-group of 4).
//   y[w,l] = sum_v AdjT[w,v] * x[v,l], order 1 and 2, per channel.
//   x slice transposed into LDS as xT[c][l][520] bf16 (520 pad: zero K tail
//   + conflict-free y1 transpose writes). y1 kept in LDS for order 2.
//   Per wave: 4 w-tiles x 4 channels; A-frag reused across channels.
//   B-frags for all 4 channels are loaded before the 4 WMMAs so a single
//   dscnt wait covers the group and the matrix ops issue back-to-back.
// ---------------------------------------------------------------------------
__global__ __launch_bounds__(256) void diffuse_kernel(
    const float* __restrict__ inflow, const float* __restrict__ outflow,
    const unsigned short* __restrict__ adjPad, unsigned short* __restrict__ hslots) {
  extern __shared__ unsigned short sm[];   // xT[4*12*520] then yT[4*12*520]
  unsigned short* xT = sm;
  unsigned short* yT = sm + 24960;

  int bid   = blockIdx.x;
  int cg    = bid % 14;
  int chain = (bid / 14) % 8;
  int n     = bid / 112;

  int matIdx;
  switch (chain) {
    case 0: matIdx = 0; break; case 1: matIdx = 1; break;
    case 2: matIdx = 2; break; case 3: matIdx = 3; break;
    case 4: matIdx = 0; break; case 5: matIdx = 1; break;
    case 6: matIdx = 4; break; default: matIdx = 5; break;
  }
  const unsigned short* Apad = adjPad + (size_t)matIdx * ADJ_EL;
  const float* xsrc = ((chain < 4) ? inflow : outflow) + (size_t)n * NSLICE + cg * 4 * CVL;
  unsigned short* dst1 = hslots + (size_t)(2 * chain)     * XSZ + (size_t)n * NSLICE + cg * 4 * CVL;
  unsigned short* dst2 = hslots + (size_t)(2 * chain + 1) * XSZ + (size_t)n * NSLICE + cg * 4 * CVL;

  int tid = threadIdx.x;
  // zero both LDS buffers (v-tail padding must be 0)
  for (int i = tid; i < 24960; i += 256) ((unsigned int*)sm)[i] = 0u;
  __syncthreads();
  // stage xT: coalesced fp32 reads, bf16 transpose into LDS
  for (int i = tid; i < 24000; i += 256) {
    int c = i / 6000, rem = i % 6000;
    int v = rem / 12, l = rem % 12;
    xT[(c * 12 + l) * 520 + v] = f2bf(xsrc[i]);
  }
  __syncthreads();

  int lane = tid & 31, wv = tid >> 5;
  int m  = lane & 15, hi = lane >> 4;
  int lcl = (m < 12) ? m : 11;             // clamp B columns 12..15 (never stored)

  for (int ord = 0; ord < 2; ++ord) {
    const unsigned short* Bsrc = ord ? yT : xT;
    unsigned short* dst = ord ? dst2 : dst1;
    for (int tp = 0; tp < 4; ++tp) {
      int t = wv * 4 + tp, w0 = t * 16;
      v8f acc[4] = {};
      // A row pointer: AdjT[w0+m][...]; 512 stride keeps b128 alignment
      const uint4v* arow = (const uint4v*)(Apad + (size_t)(w0 + m) * 512 + hi * 8);
      for (int ks = 0; ks < 16; ++ks) {
        FragB16 A;
        A.q[0] = arow[ks * 4 + 0];          // K = v0 + hi*8 .. +8
        A.q[1] = arow[ks * 4 + 2];          // K = v0 + 16 + hi*8 .. +8
        FragB16 Bf[4];
#pragma unroll
        for (int c = 0; c < 4; ++c) {       // batch all LDS loads first
          const uint4v* bp = (const uint4v*)(Bsrc + (c * 12 + lcl) * 520 + ks * 32 + hi * 8);
          Bf[c].q[0] = bp[0];                // K = v0 + hi*8 .. +8, then v0 + 16 + hi*8 .. +8
          Bf[c].q[1] = bp[2];
        }
#pragma unroll
        for (int c = 0; c < 4; ++c) {       // then 4 back-to-back WMMAs
          acc[c] = __builtin_amdgcn_wmma_f32_16x16x32_bf16(
              false, A.v, false, Bf[c].v, (short)0, acc[c], false, false);
        }
      }
#pragma unroll
      for (int c = 0; c < 4; ++c) {
#pragma unroll
        for (int r = 0; r < 8; ++r) {
          int w = w0 + r + hi * 8;           // C/D layout: VGPR r -> M = r (+8 hi)
          if (w < 500 && m < 12) {
            unsigned short bv = f2bf(acc[c][r]);
            dst[c * CVL + w * 12 + m] = bv;
            if (ord == 0) yT[(c * 12 + m) * 520 + w] = bv;
          }
        }
      }
    }
    __syncthreads();   // y1 complete before order-2 reads it
  }
}

// ---------------------------------------------------------------------------
// contract: workgroup = (n, output, 128-wide wl strip).
//   out[o, wl] = sum over 7 blocks (56 ch each, padded to 64) of W @ h,
//   + bias + fp32 residual. h tile staged transposed into LDS [128][72].
//   8 waves = 4 o-tiles x 2 wl halves; each wave 4 accumulators.
//   B-frags for the 4 wl-tiles batched ahead of the 4 WMMAs.
// ---------------------------------------------------------------------------
__global__ __launch_bounds__(256) void contract_kernel(
    const unsigned short* __restrict__ xbf, const unsigned short* __restrict__ hslots,
    const unsigned short* __restrict__ Wpad,
    const float* __restrict__ b_ii, const float* __restrict__ b_io,
    const float* __restrict__ b_oo, const float* __restrict__ b_oi,
    const float* __restrict__ inflow, const float* __restrict__ outflow,
    float* __restrict__ out) {
  __shared__ unsigned short ldsB[128 * 72];

  int bid   = blockIdx.x;
  int strip = bid % 47;
  int outi  = (bid / 47) % 4;
  int n     = bid / 188;
  int wl0   = strip * 128;

  const unsigned short* hb[7];
  {
    hb[0] = xbf + (size_t)((outi < 2) ? 0 : 1) * XSZ + (size_t)n * NSLICE;
    int base = (outi < 2) ? 0 : 8;
    int s45  = (outi & 1) ? 6 : 4;
#pragma unroll
    for (int j = 1; j <= 4; ++j)
      hb[j] = hslots + (size_t)(base + j - 1) * XSZ + (size_t)n * NSLICE;
    hb[5] = hslots + (size_t)(base + s45)     * XSZ + (size_t)n * NSLICE;
    hb[6] = hslots + (size_t)(base + s45 + 1) * XSZ + (size_t)n * NSLICE;
  }
  const unsigned short* Wp = Wpad + (size_t)outi * WPAD_EL;
  const float* bias = (outi == 0) ? b_ii : ((outi == 1) ? b_io : ((outi == 2) ? b_oo : b_oi));
  const float* res  = ((outi < 2) ? inflow : outflow) + (size_t)n * NSLICE;
  float* dst        = out + (size_t)outi * XSZ + (size_t)n * NSLICE;

  int tid = threadIdx.x;
  // zero pad channels 56..71 once (each block only rewrites c<56)
  for (int i = tid; i < 128 * 16; i += 256) {
    int wl = i >> 4, c = 56 + (i & 15);
    ldsB[wl * 72 + c] = 0;
  }

  int lane = tid & 31, wv = tid >> 5;
  int m = lane & 15, hi = lane >> 4;
  int ot  = wv >> 1;            // o-tile 0..3
  int o0  = ot * 16;
  int wtb = (wv & 1) * 4;       // this wave's first local wl-tile

  v8f acc[4] = {};

  for (int blk = 0; blk < 7; ++blk) {
    __syncthreads();
    const unsigned short* src = hb[blk];
    for (int i = tid; i < 56 * 128; i += 256) {   // coalesced read, LDS transpose
      int c = i >> 7, wl = i & 127;
      int g = wl0 + wl;
      unsigned short v2 = (g < CVL) ? src[c * CVL + g] : (unsigned short)0;
      ldsB[wl * 72 + c] = v2;
    }
    __syncthreads();
#pragma unroll
    for (int s = 0; s < 2; ++s) {                 // K = 64 per block (padded)
      FragB16 A;
      const uint4v* ap = (const uint4v*)(Wp + (o0 + m) * 512 + blk * 64 + s * 32 + hi * 8);
      A.q[0] = ap[0];
      A.q[1] = ap[2];
      FragB16 Bf[4];
#pragma unroll
      for (int q = 0; q < 4; ++q) {               // batch LDS loads
        int wlLoc = (wtb + q) * 16 + m;
        const uint4v* bp = (const uint4v*)(ldsB + wlLoc * 72 + s * 32 + hi * 8);
        Bf[q].q[0] = bp[0];
        Bf[q].q[1] = bp[2];
      }
#pragma unroll
      for (int q = 0; q < 4; ++q) {               // then 4 back-to-back WMMAs
        acc[q] = __builtin_amdgcn_wmma_f32_16x16x32_bf16(
            false, A.v, false, Bf[q].v, (short)0, acc[q], false, false);
      }
    }
  }

#pragma unroll
  for (int q = 0; q < 4; ++q) {
    int wl = wl0 + (wtb + q) * 16 + m;
#pragma unroll
    for (int r = 0; r < 8; ++r) {
      int o = o0 + r + hi * 8;
      if (o < 56 && wl < CVL)
        dst[o * CVL + wl] = acc[q][r] + bias[o] + res[o * CVL + wl];
    }
  }
}

// ---------------------------------------------------------------------------
extern "C" void kernel_launch(void* const* d_in, const int* in_sizes, int n_in,
                              void* d_out, int out_size, void* d_ws, size_t ws_size,
                              hipStream_t stream) {
  const float* inflow  = (const float*)d_in[0];
  const float* outflow = (const float*)d_in[1];
  const float* adp0    = (const float*)d_in[2];
  const float* adp1    = (const float*)d_in[3];
  const float* ii_adp  = (const float*)d_in[4];
  const float* io_adp  = (const float*)d_in[5];
  const float* oo_adp  = (const float*)d_in[6];
  const float* w_ii    = (const float*)d_in[7];
  const float* b_ii    = (const float*)d_in[8];
  const float* w_oo    = (const float*)d_in[9];
  const float* b_oo    = (const float*)d_in[10];
  const float* w_io    = (const float*)d_in[11];
  const float* b_io    = (const float*)d_in[12];
  const float* w_oi    = (const float*)d_in[13];
  const float* b_oi    = (const float*)d_in[14];

  char* ws = (char*)d_ws;
  unsigned short* adjPad = (unsigned short*)(ws);
  unsigned short* Wpad   = (unsigned short*)(ws + 3145728);
  unsigned short* xbf    = (unsigned short*)(ws + 3407872);
  unsigned short* hslots = (unsigned short*)(ws + 89423872);

  prep_adj<<<6144, 256, 0, stream>>>(adp0, adp1, ii_adp, io_adp, oo_adp, adjPad);
  prep_w<<<512, 256, 0, stream>>>(w_ii, w_io, w_oo, w_oi, Wpad);
  prep_x<<<168000, 256, 0, stream>>>(inflow, outflow, xbf);
  diffuse_kernel<<<7168, 256, 99840, stream>>>(inflow, outflow, adjPad, hslots);
  contract_kernel<<<12032, 256, 0, stream>>>(xbf, hslots, Wpad,
                                             b_ii, b_io, b_oo, b_oi,
                                             inflow, outflow, (float*)d_out);
}
